// Model_12807592476709
// MI455X (gfx1250) — matrix operand order repaired and checked by reading, not compiled, not run on hardware
//
#include <hip/hip_runtime.h>

#define BATCH 8
#define NPTS  2048
#define KNN   20
#define BN    (BATCH * NPTS)        // 16384
#define NCOLS2 (BN * KNN)           // 327680
#define EPSV  1e-5f
#define MT    4                     // M-tiles (16 rows each) per wave in GEMM

typedef __attribute__((ext_vector_type(16))) __bf16 v16bf;
typedef __attribute__((ext_vector_type(8)))  float  v8f;

// ---------- helpers ----------
__device__ __forceinline__ unsigned short f2bf(float f) {
  union { float f; unsigned int u; } c; c.f = f;
  unsigned int u = c.u;
  unsigned int r = u + 0x7FFFu + ((u >> 16) & 1u);   // round-to-nearest-even
  return (unsigned short)(r >> 16);
}
__device__ __forceinline__ float bf2f(unsigned short h) {
  union { unsigned int u; float f; } c; c.u = ((unsigned int)h) << 16;
  return c.f;
}
__device__ __forceinline__ float lrelu(float y) { return (y > 0.f) ? y : 0.2f * y; }

// ---------- kNN: top-20 by squared distance on first 3 channels ----------
// addr(b,c,n) = pos + b*bs + c*cs + n
__global__ __launch_bounds__(256)
void knn_kernel(const float* __restrict__ pos, int bs, int cs, int* __restrict__ idx_out) {
  __shared__ float sx[NPTS], sy[NPTS], sz[NPTS];
  int b = blockIdx.y;
  const float* p = pos + (size_t)b * bs;
  for (int i = threadIdx.x; i < NPTS; i += blockDim.x) {
    sx[i] = p[i];
    sy[i] = p[cs + i];
    sz[i] = p[2 * cs + i];
  }
  __syncthreads();
  int n = blockIdx.x * blockDim.x + threadIdx.x;
  float qx = sx[n], qy = sy[n], qz = sz[n];
  float dist[KNN]; int idx[KNN];
#pragma unroll
  for (int j = 0; j < KNN; ++j) { dist[j] = 3.4e38f; idx[j] = 0; }
  for (int m = 0; m < NPTS; ++m) {
    float dx = sx[m] - qx, dy = sy[m] - qy, dz = sz[m] - qz;
    float d = dx * dx + dy * dy + dz * dz;
    if (d < dist[KNN - 1]) {
      dist[KNN - 1] = d; idx[KNN - 1] = m;
#pragma unroll
      for (int j = KNN - 1; j > 0; --j) {
        if (dist[j] < dist[j - 1]) {
          float td = dist[j]; dist[j] = dist[j - 1]; dist[j - 1] = td;
          int   ti = idx[j];  idx[j]  = idx[j - 1];  idx[j - 1]  = ti;
        }
      }
    }
  }
  int* o = idx_out + ((size_t)b * NPTS + n) * KNN;
#pragma unroll
  for (int j = 0; j < KNN; ++j) o[j] = idx[j];
}

// ---------- edge features, layer 1: x (B,6,N) -> bf16 [NCOLS2][32] (12 real + pad) ----------
__global__ __launch_bounds__(256)
void edge1_kernel(const float* __restrict__ x, const int* __restrict__ idx,
                  unsigned short* __restrict__ Xe) {
  int col = blockIdx.x * blockDim.x + threadIdx.x;
  if (col >= NCOLS2) return;
  int bn = col / KNN;
  int b = bn / NPTS, n = bn % NPTS;
  int j = idx[col];
  const float* xb = x + (size_t)b * 6 * NPTS;
  unsigned short* o = Xe + (size_t)col * 32;
#pragma unroll
  for (int c = 0; c < 6; ++c) {
    float ctr = xb[c * NPTS + n];
    float gat = xb[c * NPTS + j];
    o[c]     = f2bf(gat - ctr);
    o[6 + c] = f2bf(ctr);
  }
#pragma unroll
  for (int c = 12; c < 32; ++c) o[c] = 0;
}

// ---------- edge features, 64-ch input xf [64][BN] -> bf16 [NCOLS2][128] ----------
__global__ __launch_bounds__(256)
void edge64_kernel(const float* __restrict__ xf, const int* __restrict__ idx,
                   unsigned short* __restrict__ Xe) {
  int col = blockIdx.x * blockDim.x + threadIdx.x;
  if (col >= NCOLS2) return;
  int bn = col / KNN;
  int b = bn / NPTS;
  int j = b * NPTS + idx[col];
  unsigned short* o = Xe + (size_t)col * 128;
#pragma unroll 8
  for (int c = 0; c < 64; ++c) {
    float ctr = xf[(size_t)c * BN + bn];
    float gat = xf[(size_t)c * BN + j];
    o[c]      = f2bf(gat - ctr);
    o[64 + c] = f2bf(ctr);
  }
}

// ---------- weight f32 [M][C] -> bf16 [M][Kpad] (zero pad) ----------
__global__ __launch_bounds__(256)
void wprep_kernel(const float* __restrict__ w, unsigned short* __restrict__ wb,
                  int M, int C, int Kpad) {
  int i = blockIdx.x * blockDim.x + threadIdx.x;
  if (i >= M * Kpad) return;
  int m = i / Kpad, k = i % Kpad;
  wb[i] = (k < C) ? f2bf(w[m * C + k]) : (unsigned short)0;
}

// ---------- WMMA GEMM: Y[M][Ncols] = Wb[M][Kpad] * Xb^T (Xb stored [Ncols][Kpad]) ----------
// one wave per 64(M) x 16(N) strip: B fragment loaded once, reused by 4 WMMAs
__global__ __launch_bounds__(256)
void gemm_wmma(const unsigned short* __restrict__ Wb,
               const unsigned short* __restrict__ Xb,
               float* __restrict__ Y,
               int M, int Kpad, int Ncols) {
  int wave = (int)(((unsigned)blockIdx.x * blockDim.x + threadIdx.x) >> 5);
  int lane = threadIdx.x & 31;
  int tiles_n = Ncols >> 4;
  int tm = wave / tiles_n;            // covers rows [tm*64, tm*64+64)
  int tn = wave % tiles_n;
  int half = lane >> 4;
  int l16  = lane & 15;

  union { unsigned int u[8]; v16bf v; } am[MT];
  union { unsigned int u[8]; v16bf v; } bm;
  v8f acc[MT];
#pragma unroll
  for (int t = 0; t < MT; ++t) acc[t] = (v8f){};

  const unsigned short* xcol = Xb + (size_t)(tn * 16 + l16) * Kpad;
  const unsigned int* wrow[MT];
#pragma unroll
  for (int t = 0; t < MT; ++t)
    wrow[t] = (const unsigned int*)(Wb + (size_t)(tm * 64 + t * 16 + l16) * Kpad);

  for (int k0 = 0; k0 < Kpad; k0 += 32) {
    const unsigned int* xp = (const unsigned int*)(xcol + k0 + 16 * half);
#pragma unroll
    for (int v = 0; v < 8; ++v) bm.u[v] = xp[v];
    int kp = (k0 >> 1) + 8 * half;   // dword index of K-pair base (contiguous k = 16h + s, matches B)
#pragma unroll
    for (int t = 0; t < MT; ++t) {
#pragma unroll
      for (int v = 0; v < 4; ++v) am[t].u[v]     = wrow[t][kp + v];
#pragma unroll
      for (int v = 0; v < 4; ++v) am[t].u[4 + v] = wrow[t][kp + 4 + v];
      acc[t] = __builtin_amdgcn_wmma_f32_16x16x32_bf16(
          false, am[t].v, false, bm.v, (short)0, acc[t], false, false);
    }
  }
#pragma unroll
  for (int t = 0; t < MT; ++t) {
    float* yb = Y + (size_t)(tm * 64 + t * 16 + 8 * half) * Ncols + tn * 16 + l16;
#pragma unroll
    for (int i = 0; i < 8; ++i) yb[(size_t)i * Ncols] = acc[t][i];
  }
}

// ---------- per-channel mean/var: one block per channel ----------
__global__ __launch_bounds__(256)
void stats_kernel(const float* __restrict__ Y, int Ncols,
                  float* __restrict__ stats, int M) {
  int c = blockIdx.x;
  const float* row = Y + (size_t)c * Ncols;
  float s = 0.f, ss = 0.f;
  for (int i = threadIdx.x; i < Ncols; i += blockDim.x) {
    float v = row[i];
    s += v; ss += v * v;
  }
  __shared__ float sh[512];
  sh[threadIdx.x] = s; sh[256 + threadIdx.x] = ss;
  __syncthreads();
  for (int w = 128; w > 0; w >>= 1) {
    if (threadIdx.x < w) {
      sh[threadIdx.x]       += sh[threadIdx.x + w];
      sh[256 + threadIdx.x] += sh[256 + threadIdx.x + w];
    }
    __syncthreads();
  }
  if (threadIdx.x == 0) {
    float inv = 1.f / (float)Ncols;
    float mean = sh[0] * inv;
    float var  = sh[256] * inv - mean * mean;
    stats[c] = mean; stats[M + c] = var;
  }
}

// ---------- BN + lrelu, write bf16 [Ncols][M] with b128 packed stores ----------
// one thread per column; per-channel scale/shift staged in LDS; M <= 512, M % 8 == 0
__global__ __launch_bounds__(256)
void bn_apply_bf16(const float* __restrict__ Y, const float* __restrict__ stats,
                   const float* __restrict__ g, const float* __restrict__ bta,
                   unsigned short* __restrict__ Out, int M, int Ncols) {
  __shared__ float ssc[512], ssh[512];
  for (int c = threadIdx.x; c < M; c += blockDim.x) {
    float sc = g[c] * rsqrtf(stats[M + c] + EPSV);
    ssc[c] = sc;
    ssh[c] = bta[c] - stats[c] * sc;
  }
  __syncthreads();
  int col = blockIdx.x * blockDim.x + threadIdx.x;
  if (col >= Ncols) return;
  for (int c0 = 0; c0 < M; c0 += 8) {
    unsigned int pk[4];
#pragma unroll
    for (int j = 0; j < 4; ++j) {
      int c = c0 + 2 * j;
      float y0 = lrelu(Y[(size_t)c * Ncols + col]       * ssc[c]     + ssh[c]);
      float y1 = lrelu(Y[(size_t)(c + 1) * Ncols + col] * ssc[c + 1] + ssh[c + 1]);
      pk[j] = (unsigned int)f2bf(y0) | ((unsigned int)f2bf(y1) << 16);
    }
    uint4 val; val.x = pk[0]; val.y = pk[1]; val.z = pk[2]; val.w = pk[3];
    *(uint4*)(Out + (size_t)col * M + c0) = val;
  }
}

// ---------- BN + lrelu + max over KNN: [64][NCOLS2] -> [64][BN] ----------
__global__ __launch_bounds__(256)
void bn_apply_maxk(const float* __restrict__ Y, const float* __restrict__ stats,
                   const float* __restrict__ g, const float* __restrict__ bta,
                   float* __restrict__ Xout) {
  int i = blockIdx.x * blockDim.x + threadIdx.x;
  if (i >= 64 * BN) return;
  int c = i / BN;
  float sc = g[c] * rsqrtf(stats[64 + c] + EPSV);
  float sh = bta[c] - stats[c] * sc;
  const float* row = Y + (size_t)c * NCOLS2 + (size_t)(i % BN) * KNN;
  float m = -3.4e38f;
#pragma unroll
  for (int j = 0; j < KNN; ++j) m = fmaxf(m, lrelu(row[j] * sc + sh));
  Xout[i] = m;
}

// ---------- BN + lrelu + max over n (per batch): [1024][BN] -> g [1024][B] ----------
__global__ __launch_bounds__(256)
void bn_apply_gmax(const float* __restrict__ Y, const float* __restrict__ stats,
                   const float* __restrict__ g, const float* __restrict__ bta,
                   float* __restrict__ Gout) {
  int i = blockIdx.x * blockDim.x + threadIdx.x;
  if (i >= 1024 * BATCH) return;
  int c = i / BATCH, b = i % BATCH;
  float sc = g[c] * rsqrtf(stats[1024 + c] + EPSV);
  float sh = bta[c] - stats[c] * sc;
  const float* row = Y + (size_t)c * BN + (size_t)b * NPTS;
  float m = -3.4e38f;
  for (int n = 0; n < NPTS; ++n) m = fmaxf(m, lrelu(row[n] * sc + sh));
  Gout[(size_t)c * BATCH + b] = m;
}

// ---------- concat x1,x2,x3 -> bf16 [BN][192] ----------
__global__ __launch_bounds__(256)
void cat_kernel(const float* __restrict__ x1, const float* __restrict__ x2,
                const float* __restrict__ x3, unsigned short* __restrict__ Out) {
  int bn = blockIdx.x * blockDim.x + threadIdx.x;
  if (bn >= BN) return;
  unsigned short* o = Out + (size_t)bn * 192;
#pragma unroll 8
  for (int c = 0; c < 64; ++c) {
    o[c]       = f2bf(x1[(size_t)c * BN + bn]);
    o[64 + c]  = f2bf(x2[(size_t)c * BN + bn]);
    o[128 + c] = f2bf(x3[(size_t)c * BN + bn]);
  }
}

// ---------- build [g broadcast, x1, x2, x3] -> bf16 [BN][1216] ----------
// g[:,b] is identical for every point of batch b: async-stage it into LDS once
// per block (GLOBAL_LOAD_ASYNC_TO_LDS_B32, ASYNCcnt-tracked), then broadcast.
__global__ __launch_bounds__(256)
void in7_kernel(const float* __restrict__ Gout, const float* __restrict__ x1,
                const float* __restrict__ x2, const float* __restrict__ x3,
                unsigned short* __restrict__ Out) {
  __shared__ float gsh[1024];
  int bn0 = blockIdx.x * blockDim.x;       // 256 cols per block, all in same batch
  int b = bn0 / NPTS;
  for (int c = threadIdx.x; c < 1024; c += blockDim.x) {
    const float* gp = Gout + (size_t)c * BATCH + b;
    unsigned int lds_off = (unsigned int)(size_t)(gsh + c);
    asm volatile("global_load_async_to_lds_b32 %0, %1, off"
                 :: "v"(lds_off), "v"(gp) : "memory");
  }
  asm volatile("s_wait_asynccnt 0x0" ::: "memory");
  __syncthreads();

  int bn = bn0 + threadIdx.x;
  unsigned short* o = Out + (size_t)bn * 1216;
#pragma unroll 4
  for (int c = 0; c < 1024; ++c) o[c] = f2bf(gsh[c]);
#pragma unroll 8
  for (int c = 0; c < 64; ++c) {
    o[1024 + c] = f2bf(x1[(size_t)c * BN + bn]);
    o[1088 + c] = f2bf(x2[(size_t)c * BN + bn]);
    o[1152 + c] = f2bf(x3[(size_t)c * BN + bn]);
  }
}

// ---------- final conv: w9 (4x256) * act [BN][256] -> out (B,4,N) f32 ----------
__global__ __launch_bounds__(256)
void final_kernel(const float* __restrict__ w9, const unsigned short* __restrict__ A,
                  float* __restrict__ out) {
  int bn = blockIdx.x * blockDim.x + threadIdx.x;
  if (bn >= BN) return;
  int b = bn / NPTS, n = bn % NPTS;
  const unsigned short* a = A + (size_t)bn * 256;
#pragma unroll
  for (int o = 0; o < 4; ++o) {
    float s = 0.f;
    for (int c = 0; c < 256; ++c) s += w9[o * 256 + c] * bf2f(a[c]);
    out[((size_t)b * 4 + o) * NPTS + n] = s;
  }
}

// =========================================================================
extern "C" void kernel_launch(void* const* d_in, const int* in_sizes, int n_in,
                              void* d_out, int out_size, void* d_ws, size_t ws_size,
                              hipStream_t stream) {
  (void)in_sizes; (void)n_in; (void)out_size; (void)ws_size;
  const float* x = (const float*)d_in[0];
  const float* w[9];
  for (int i = 0; i < 9; ++i) w[i] = (const float*)d_in[1 + i];
  const float* gg[8]; const float* bb[8];
  for (int i = 0; i < 8; ++i) {
    gg[i] = (const float*)d_in[10 + 2 * i];
    bb[i] = (const float*)d_in[11 + 2 * i];
  }
  float* out = (float*)d_out;

  char* ws = (char*)d_ws;
  size_t off = 0;
  auto alloc = [&](size_t bytes) -> char* {
    char* p = ws + off;
    off += (bytes + 255) & ~(size_t)255;
    return p;
  };
  int*   idx   = (int*)  alloc((size_t)NCOLS2 * 4);
  float* stats = (float*)alloc(2 * 1024 * 4);
  float* x1    = (float*)alloc((size_t)64 * BN * 4);
  float* x2    = (float*)alloc((size_t)64 * BN * 4);
  float* x3    = (float*)alloc((size_t)64 * BN * 4);
  float* gmx   = (float*)alloc((size_t)1024 * BATCH * 4);
  static const int wM[8] = {64, 64, 64, 64, 64, 1024, 512, 256};
  static const int wC[8] = {12, 64, 128, 64, 128, 192, 1216, 512};
  static const int wK[8] = {32, 64, 128, 64, 128, 192, 1216, 512};
  unsigned short* wb[8];
  for (int i = 0; i < 8; ++i) wb[i] = (unsigned short*)alloc((size_t)wM[i] * wK[i] * 2);
  unsigned short* bufA = (unsigned short*)alloc((size_t)NCOLS2 * 128 * 2); // 80 MB
  float*          bufB = (float*)         alloc((size_t)64 * NCOLS2 * 4);  // 80 MB

  auto gemm = [&](int i, const unsigned short* Xb, int Ncols) {
    int waves = (wM[i] / (16 * MT)) * (Ncols / 16);
    gemm_wmma<<<waves / 8, 256, 0, stream>>>(wb[i], Xb, bufB, wM[i], wK[i], Ncols);
    stats_kernel<<<wM[i], 256, 0, stream>>>(bufB, Ncols, stats, wM[i]);
  };

  // weight prep (bf16, pad K to 32-multiple)
  for (int i = 0; i < 8; ++i) {
    int tot = wM[i] * wK[i];
    wprep_kernel<<<(tot + 255) / 256, 256, 0, stream>>>(w[i], wb[i], wM[i], wC[i], wK[i]);
  }

  dim3 kgrid(NPTS / 256, BATCH);

  // ---- stage 1 ----
  knn_kernel<<<kgrid, 256, 0, stream>>>(x, 6 * NPTS, NPTS, idx);
  edge1_kernel<<<(NCOLS2 + 255) / 256, 256, 0, stream>>>(x, idx, bufA);
  gemm(0, bufA, NCOLS2);
  bn_apply_bf16<<<(NCOLS2 + 255) / 256, 256, 0, stream>>>(bufB, stats, gg[0], bb[0], bufA, 64, NCOLS2);
  gemm(1, bufA, NCOLS2);
  bn_apply_maxk<<<(64 * BN + 255) / 256, 256, 0, stream>>>(bufB, stats, gg[1], bb[1], x1);

  // ---- stage 2 ----
  knn_kernel<<<kgrid, 256, 0, stream>>>(x1, NPTS, BN, idx);
  edge64_kernel<<<(NCOLS2 + 255) / 256, 256, 0, stream>>>(x1, idx, bufA);
  gemm(2, bufA, NCOLS2);
  bn_apply_bf16<<<(NCOLS2 + 255) / 256, 256, 0, stream>>>(bufB, stats, gg[2], bb[2], bufA, 64, NCOLS2);
  gemm(3, bufA, NCOLS2);
  bn_apply_maxk<<<(64 * BN + 255) / 256, 256, 0, stream>>>(bufB, stats, gg[3], bb[3], x2);

  // ---- stage 3 ----
  knn_kernel<<<kgrid, 256, 0, stream>>>(x2, NPTS, BN, idx);
  edge64_kernel<<<(NCOLS2 + 255) / 256, 256, 0, stream>>>(x2, idx, bufA);
  gemm(4, bufA, NCOLS2);
  bn_apply_maxk<<<(64 * BN + 255) / 256, 256, 0, stream>>>(bufB, stats, gg[4], bb[4], x3);

  // ---- global stage ----
  cat_kernel<<<(BN + 255) / 256, 256, 0, stream>>>(x1, x2, x3, bufA);
  gemm(5, bufA, BN);
  bn_apply_gmax<<<(1024 * BATCH + 255) / 256, 256, 0, stream>>>(bufB, stats, gg[5], bb[5], gmx);
  in7_kernel<<<BN / 256, 256, 0, stream>>>(gmx, x1, x2, x3, bufA);
  gemm(6, bufA, BN);
  bn_apply_bf16<<<(BN + 255) / 256, 256, 0, stream>>>(bufB, stats, gg[6], bb[6], bufA, 512, BN);
  gemm(7, bufA, BN);
  bn_apply_bf16<<<(BN + 255) / 256, 256, 0, stream>>>(bufB, stats, gg[7], bb[7], bufA, 256, BN);
  final_kernel<<<(BN + 255) / 256, 256, 0, stream>>>(w[8], bufA, out);
}
